// ConvXD_42932493091522
// MI455X (gfx1250) — matrix operand order repaired and checked by reading, not compiled, not run on hardware
//
#include <hip/hip_runtime.h>
#include <hip/hip_bf16.h>

// ---------------------------------------------------------------------------
// 4-D conv (B=8, Cin=8, Cout=16, 3^4 taps, 24^4) as LDS-staged implicit GEMM
// on CDNA5 WMMA: D(16x16 f32) += A(16x32 f16) x B(32x16 f16), K=648 -> 21x32.
//  Pass 1: zero-pad + fp32->f16 convert into d_ws (8,8,26^4) f16 (~58 MB).
//  Pass 2: one block per (b,s1,s2): stage the 8x3x3x26x26 receptive field in
//  LDS (97 KB f16); 12 waves x 3 register-blocked tiles; dynamic K loop so
//  nothing spills; offset/B-fragment loads amortized over 3 WMMAs.
// ---------------------------------------------------------------------------

typedef __attribute__((ext_vector_type(16))) _Float16 v16h;
typedef __attribute__((ext_vector_type(8)))  float    v8f;
typedef __attribute__((ext_vector_type(4)))  float    v4f;
typedef unsigned long long u64t;

// ---- geometry ----
#define CIN      8
#define COUT     16
#define SDIM     24
#define PDIM     26
#define KTOT     648                        // 81 taps * 8 channels
#define KCHUNKS  21                         // ceil(648/32)
#define PPLANE   (PDIM*PDIM)                // 676 (p3,p4) plane, contiguous
#define PSTR1    (PDIM*PPLANE)              // 17576
#define PSTRC    (PDIM*PSTR1)               // 456976
#define NPADH    (8*CIN*PSTRC)              // 29,246,464 halfs (~58 MB)
#define MSLICE   (SDIM*SDIM)                // 576
#define OSLICE   (SDIM*SDIM*MSLICE)         // 331776 per-channel volume
// LDS stage strides (halfs): (c, d1, d2, p3, p4)
#define LSTR3    PDIM                       // 26
#define LSTR2    PPLANE                     // 676
#define LSTR1    (3*PPLANE)                 // 2028
#define LSTRC    (9*PPLANE)                 // 6084
#define STAGEH   (CIN*LSTRC)                // 48672 halfs = 97,344 B
#define FILLQ    (CIN*9*169)                // 12168 8-byte copies per block
#define NTHREADS 384
#define NWAVES   (NTHREADS/32)              // 12
#define TPW      3                          // tiles per wave (12*3 = 36)

// ---------------------------------------------------------------------------
// Pass 1: zero-padded f32 -> f16 copy  x(8,8,24^4) -> xpadh(8,8,26^4)
// ---------------------------------------------------------------------------
__global__ __launch_bounds__(256) void conv4d_pad_f16_kernel(
    const float* __restrict__ x, _Float16* __restrict__ xpadh, int n)
{
    int i = blockIdx.x * 256 + threadIdx.x;
    if (i >= n) return;
    int c4 = i % PDIM;  int t = i / PDIM;
    int c3 = t % PDIM;  t /= PDIM;
    int c2 = t % PDIM;  t /= PDIM;
    int c1 = t % PDIM;  t /= PDIM;          // t = b*CIN + c
    _Float16 v = (_Float16)0.f;
    if (c1 >= 1 && c1 <= SDIM && c2 >= 1 && c2 <= SDIM &&
        c3 >= 1 && c3 <= SDIM && c4 >= 1 && c4 <= SDIM) {
        v = (_Float16)__builtin_nontemporal_load(
            &x[(((t*SDIM + (c1-1))*SDIM + (c2-1))*SDIM + (c3-1))*SDIM + (c4-1)]);
    }
    xpadh[i] = v;    // regular store: immediately reused, keep in L2
}

// ---------------------------------------------------------------------------
// Pass 2: one block per (b,s1,s2); 36 tiles from an LDS-resident stage.
// ---------------------------------------------------------------------------
__global__ __launch_bounds__(NTHREADS) void conv4d_wmma_lds_kernel(
    const _Float16* __restrict__ xpadh,
    const float*    __restrict__ weight,   // (16, 648) kernel-major/chan-minor
    const float*    __restrict__ bias,     // (16)
    float*          __restrict__ out)      // (8,16,24,24,24,24)
{
    __shared__ u64t stageQ[STAGEH / 4];          // 97,344 B, 8B-aligned
    __shared__ v16h wfrag[KCHUNKS * 32];         // B fragments (contiguous K order)
    __shared__ int  koffB[KCHUNKS * 32];         // LDS byte offset per logical K

    const int tid = threadIdx.x;

    // block -> (b, s1, s2)
    int blk = blockIdx.x;
    int s2  = blk % SDIM;  int bt = blk / SDIM;
    int s1  = bt % SDIM;
    int b   = bt / SDIM;

    // ---- prep: koff table (BYTES into LDS stage) -------------------------
    for (int kk = tid; kk < KCHUNKS * 32; kk += NTHREADS) {
        int off = 0;
        if (kk < KTOT) {
            int c  = kk & 7;
            int tp = kk >> 3;               // tap, kernel-major
            int k4 = tp % 3;
            int k3 = (tp / 3) % 3;
            int k2 = (tp / 9) % 3;
            int k1 = tp / 27;
            off = (c*LSTRC + k1*LSTR1 + k2*LSTR2 + k3*LSTR3 + k4) * 2;
        }
        koffB[kk] = off;                    // kk>=648: weight is zero there
    }
    // ---- prep: weight -> f16 B fragments (contiguous K: k = 16*half + h) -----
    _Float16* wf = (_Float16*)wfrag;
    for (int i = tid; i < KCHUNKS * 32 * 16; i += NTHREADS) {
        int kc   = i >> 9;
        int r    = i & 511;
        int lane = r >> 4;
        int h    = r & 15;
        int n    = lane & 15;
        int p    = (lane & 16) + h;         // physical K slot
        int jo   = p;
        int kk   = kc * 32 + jo;            // logical K
        _Float16 v = (_Float16)0.f;
        if (kk < KTOT) v = (_Float16)weight[n * KTOT + kk];
        wf[i] = v;
    }
    // ---- stage fill: 8-byte copies, branch-free --------------------------
    {
        const u64t* src = (const u64t*)xpadh;
        for (int i = tid; i < FILLQ; i += NTHREADS) {
            int plane = i / 169;
            int q     = i - plane * 169;
            int c     = plane / 9;
            int d     = plane - c * 9;
            int d1    = d / 3;
            int d2    = d - d1 * 3;
            int g = ((((b*CIN + c) * PDIM + (s1 + d1)) * PDIM + (s2 + d2)) * PPLANE) >> 2;
            stageQ[plane * 169 + q] = src[g + q];
        }
    }
    __syncthreads();

    const int lane  = tid & 31;
    const int wave  = tid >> 5;
    const int hioff = (lane & 16) ? 16 : 0;      // A-side logical-K shift
    const char* stg = (const char*)stageQ;

    // 3 register-blocked tiles per wave: mbase = wave*48 + i*16
    unsigned lB[TPW];
#pragma unroll
    for (int i = 0; i < TPW; ++i) {
        int m  = wave * (TPW * 16) + i * 16 + (lane & 15);
        int s3 = m / SDIM;
        int s4 = m % SDIM;
        lB[i]  = (unsigned)((s3 * LSTR3 + s4) * 2);
    }

    v8f acc[TPW];
#pragma unroll
    for (int i = 0; i < TPW; ++i) acc[i] = (v8f){};

#pragma unroll 1   // keep K loop rolled: no hoisting, no spills
    for (int kc = 0; kc < KCHUNKS; ++kc) {
        const int kbase = kc * 32 + hioff;
        int ko[16];
#pragma unroll
        for (int h = 0; h < 16; ++h) ko[h] = koffB[kbase + h];

        v16h bfr = wfrag[kc * 32 + lane];

#pragma unroll
        for (int i = 0; i < TPW; ++i) {
            // 16 independent zero-extended u16 LDS gathers, then pair-pack
            unsigned e[16];
#pragma unroll
            for (int h = 0; h < 16; ++h)
                e[h] = *(const unsigned short*)(stg + (lB[i] + (unsigned)ko[h]));

            union { unsigned u[8]; v16h h; } af;
#pragma unroll
            for (int p = 0; p < 8; ++p)
                af.u[p] = e[2*p] | (e[2*p + 1] << 16);

            acc[i] = __builtin_amdgcn_wmma_f32_16x16x32_f16(
                         false, af.h, false, bfr, (short)0, acc[i], false, false);
        }
    }

    // ---- store: C rows contiguous along flattened (s3,s4) ----------------
    const float bv = bias[lane & 15];
    int o = lane & 15;
#pragma unroll
    for (int i = 0; i < TPW; ++i) {
        int mbase = wave * (TPW * 16) + i * 16;
        int obase = __builtin_amdgcn_readfirstlane(
                        (((b * COUT) * SDIM + s1) * SDIM + s2) * MSLICE + mbase);
        char* so  = (char*)(out + obase);
        unsigned ovoff = (unsigned)(o * OSLICE + ((lane & 16) ? 8 : 0)) * 4u;

        v4f lo, hi;
#pragma unroll
        for (int r = 0; r < 4; ++r) { lo[r] = acc[i][r] + bv; hi[r] = acc[i][r + 4] + bv; }
        __builtin_nontemporal_store(lo, (v4f*)(so + ovoff));
        __builtin_nontemporal_store(hi, (v4f*)(so + ovoff + 16));
    }
}

// ---------------------------------------------------------------------------
extern "C" void kernel_launch(void* const* d_in, const int* in_sizes, int n_in,
                              void* d_out, int out_size, void* d_ws, size_t ws_size,
                              hipStream_t stream) {
    const float* x      = (const float*)d_in[0];
    const float* weight = (const float*)d_in[1];
    const float* bias   = (const float*)d_in[2];
    float*       out    = (float*)d_out;
    _Float16*    xpadh  = (_Float16*)d_ws;       // needs NPADH*2 = ~58 MB

    int npad = NPADH;
    conv4d_pad_f16_kernel<<<(npad + 255) / 256, 256, 0, stream>>>(x, xpadh, npad);

    int nblocks = 8 * SDIM * SDIM;               // 4608 blocks, 384 threads
    conv4d_wmma_lds_kernel<<<nblocks, NTHREADS, 0, stream>>>(
        xpadh, weight, bias, out);
}
